// RGRNN_43250320671336
// MI455X (gfx1250) — compile-verified, matrix operand order checked by reading, not run on hardware
//
#include <hip/hip_runtime.h>
#include <hip/hip_bf16.h>

// ---------------------------------------------------------------------------
// RGRNN on MI455X (gfx1250): bf16 WMMA GEMMs with f32 accumulation.
//   B=64, T=512, I=1024, H=1024
// Phase 0: convert x / weights to bf16, zero h0.
// Phase 1: Xp = x @ W_ih^T + b_ih + b_hh  (parallel 32768x1024x1024 WMMA GEMM)
// Phase 2: per t: K_A: h_rnn = tanh(Xp[:,t,:] + h @ W_hh^T)        (WMMA)
//                 K_B: stats = h_rnn @ W_g^T + b_g; sample/mu/std  (WMMA x2)
// Phase 3: deterministic reduction of per-tile log-prob partials.
// ---------------------------------------------------------------------------

#define Bb 64
#define Tt 512
#define Ii 1024
#define Hh 1024

typedef __attribute__((ext_vector_type(16))) __bf16 v16bf;
typedef __attribute__((ext_vector_type(8)))  float  v8f;

static constexpr size_t SEQ_N    = (size_t)Bb * Tt * Hh;      // 33554432
static constexpr size_t PROB_OFF = SEQ_N;                     // h_probs (B,T)
static constexpr size_t MUS_OFF  = SEQ_N + (size_t)Bb * Tt;   // 33587200
static constexpr size_t STDS_OFF = MUS_OFF + SEQ_N;           // 67141632

union FragU {
    v16bf v;
    uint4 q[2];
};

// Load a 16x32 bf16 fragment (A layout; also used for B by loading W rows,
// since W is stored (N,K) row-major and B columns == W rows).
// ISA layout: lanes 0-15 hold rows, K-halves {0..7,16..23} / {8..15,24..31}.
__device__ __forceinline__ v16bf load_frag(const __hip_bfloat16* base, int ld,
                                           int row0, int k0, int lane) {
    const int r     = row0 + (lane & 15);
    const int khalf = (lane >> 4) << 3;              // 0 or 8
    const __hip_bfloat16* p = base + (size_t)r * ld + k0 + khalf;
    FragU f;
    f.q[0] = *(const uint4*)(p);                     // k0+khalf   .. +7
    f.q[1] = *(const uint4*)(p + 16);                // k0+16+khalf.. +7
    return f.v;
}

__device__ __forceinline__ float softplusf(float x) {
    // stable: max(x,0) + log1p(exp(-|x|))
    return fmaxf(x, 0.0f) + log1pf(expf(-fabsf(x)));
}

// ------------------------------ Phase 0 ------------------------------------
__global__ void k_f32_to_bf16(const float* __restrict__ src,
                              __hip_bfloat16* __restrict__ dst, int n) {
    int i      = blockIdx.x * blockDim.x + threadIdx.x;
    int stride = gridDim.x * blockDim.x;
    for (; i < n; i += stride) dst[i] = __float2bfloat16(src[i]);
}

__global__ void k_zero_bf16(__hip_bfloat16* __restrict__ dst, int n) {
    int i      = blockIdx.x * blockDim.x + threadIdx.x;
    int stride = gridDim.x * blockDim.x;
    for (; i < n; i += stride) dst[i] = __float2bfloat16(0.0f);
}

// ------------------------------ Phase 1 ------------------------------------
// Xp[(b*T+t), n] = sum_k x[(b*T+t), k] * w_ih[n, k] + b_ih[n] + b_hh[n]
// grid (64 n-tiles, 512), block 128 (4 waves); wave = one 16x16 output tile.
__global__ void k_xp_gemm(const __hip_bfloat16* __restrict__ xbf,
                          const __hip_bfloat16* __restrict__ wih,
                          const float* __restrict__ b_ih,
                          const float* __restrict__ b_hh,
                          float* __restrict__ Xp) {
    const int lane = threadIdx.x & 31;
    const int wave = threadIdx.x >> 5;
    const int col0 = blockIdx.x * 16;
    const int row0 = (blockIdx.y * 4 + wave) * 16;

    v8f c = {};
    for (int k = 0; k < Ii; k += 32) {
        v16bf a = load_frag(xbf, Ii, row0, k, lane);
        v16bf b = load_frag(wih, Ii, col0, k, lane);
        c = __builtin_amdgcn_wmma_f32_16x16x32_bf16(false, a, false, b,
                                                    (short)0, c, false, false);
    }
    const int hi = lane >> 4, n = lane & 15;
    const int col = col0 + n;
    const float bias = b_ih[col] + b_hh[col];
#pragma unroll
    for (int v = 0; v < 8; ++v) {
        const int m = row0 + v + 8 * hi;
        Xp[(size_t)m * Hh + col] = c[v] + bias;
    }
}

// ------------------------------ Phase 2a -----------------------------------
// h_rnn = tanh(Xp[b*T+t] + h @ W_hh^T).  grid 64 (n-tiles), block 128.
__global__ void k_rnn_step(const __hip_bfloat16* __restrict__ h,
                           const __hip_bfloat16* __restrict__ whh,
                           const float* __restrict__ Xp,
                           __hip_bfloat16* __restrict__ hrnn, int t) {
    const int lane = threadIdx.x & 31;
    const int wave = threadIdx.x >> 5;
    const int col0 = blockIdx.x * 16;
    const int row0 = wave * 16;                       // batch rows 0..63

    v8f c = {};
    for (int k = 0; k < Hh; k += 32) {
        v16bf a = load_frag(h,   Hh, row0, k, lane);
        v16bf b = load_frag(whh, Hh, col0, k, lane);
        c = __builtin_amdgcn_wmma_f32_16x16x32_bf16(false, a, false, b,
                                                    (short)0, c, false, false);
    }
    const int hi = lane >> 4, n = lane & 15;
    const int col = col0 + n;
#pragma unroll
    for (int v = 0; v < 8; ++v) {
        const int bidx = row0 + v + 8 * hi;           // batch index 0..63
        const float xv = c[v] + Xp[((size_t)bidx * Tt + t) * Hh + col];
        hrnn[bidx * Hh + col] = __float2bfloat16(tanhf(xv));
    }
}

// ------------------------------ Phase 2b -----------------------------------
// stats = h_rnn @ W_g^T + b_g; std/mu/sample; per-tile log-prob partials.
// Each wave computes the paired std (cols n) and mu (cols n+H) 16x16 tiles.
__global__ void k_gauss_step(const __hip_bfloat16* __restrict__ hrnn,
                             const __hip_bfloat16* __restrict__ wg,
                             const float* __restrict__ b_g,
                             const float* __restrict__ eps,
                             float* __restrict__ out,
                             __hip_bfloat16* __restrict__ hnext,
                             float* __restrict__ partials, int t) {
    const int lane  = threadIdx.x & 31;
    const int wave  = threadIdx.x >> 5;
    const int ntile = blockIdx.x;                     // 0..63
    const int col0  = ntile * 16;
    const int row0  = wave * 16;                      // batch rows 0..63

    v8f cs = {};
    v8f cm = {};
    for (int k = 0; k < Hh; k += 32) {
        v16bf a  = load_frag(hrnn, Hh, row0, k, lane);
        v16bf bs = load_frag(wg, Hh, col0, k, lane);        // rows n      -> std
        v16bf bm = load_frag(wg, Hh, Hh + col0, k, lane);   // rows H+n    -> mu
        cs = __builtin_amdgcn_wmma_f32_16x16x32_bf16(false, a, false, bs,
                                                     (short)0, cs, false, false);
        cm = __builtin_amdgcn_wmma_f32_16x16x32_bf16(false, a, false, bm,
                                                     (short)0, cm, false, false);
    }
    const int hi = lane >> 4, n = lane & 15;
    const int col = col0 + n;
    const float bg_s = b_g[col];
    const float bg_m = b_g[Hh + col];

    float pv[8];
#pragma unroll
    for (int v = 0; v < 8; ++v) {
        const int bidx = row0 + v + 8 * hi;
        const size_t g = ((size_t)bidx * Tt + t) * Hh + col;
        const float sd  = softplusf(cs[v] + bg_s);
        const float mu  = cm[v] + bg_m;
        const float e   = eps[g];
        const float smp = mu + sd * e;
        out[g]            = smp;                      // h_seq
        out[MUS_OFF + g]  = mu;                       // h_mus
        out[STDS_OFF + g] = sd;                       // h_stds
        hnext[bidx * Hh + col] = __float2bfloat16(smp);
        pv[v] = -0.5f * e * e - logf(sd);             // z == eps exactly
    }
    // Row-wise sums: row m lives in lanes of one 16-lane half; reduce across it.
#pragma unroll
    for (int v = 0; v < 8; ++v) {
#pragma unroll
        for (int off = 1; off < 16; off <<= 1)
            pv[v] += __shfl_xor(pv[v], off, 16);
    }
    if (n == 0) {
#pragma unroll
        for (int v = 0; v < 8; ++v) {
            const int bidx = row0 + v + 8 * hi;
            partials[((size_t)t * Bb + bidx) * 64 + ntile] = pv[v];
        }
    }
}

// ------------------------------ Phase 3 ------------------------------------
__global__ void k_reduce_prob(const float* __restrict__ partials,
                              float* __restrict__ out) {
    const int idx = blockIdx.x * blockDim.x + threadIdx.x;  // t*64 + b
    if (idx >= Tt * Bb) return;
    const int t = idx >> 6;
    const int b = idx & 63;
    float s = -0.5f * 1.8378770664093453f * (float)Hh;      // -0.5*H*log(2pi)
    const float* p = partials + (size_t)idx * 64;
#pragma unroll 8
    for (int j = 0; j < 64; ++j) s += p[j];
    out[PROB_OFF + (size_t)b * Tt + t] = s;                 // h_probs (B,T)
}

// ---------------------------------------------------------------------------
extern "C" void kernel_launch(void* const* d_in, const int* in_sizes, int n_in,
                              void* d_out, int out_size, void* d_ws, size_t ws_size,
                              hipStream_t stream) {
    const float* x    = (const float*)d_in[0];   // (B,T,I)
    const float* eps  = (const float*)d_in[1];   // (B,T,H)
    const float* w_ih = (const float*)d_in[2];   // (H,I)
    const float* w_hh = (const float*)d_in[3];   // (H,H)
    const float* b_ih = (const float*)d_in[4];   // (H,)
    const float* b_hh = (const float*)d_in[5];   // (H,)
    const float* w_g  = (const float*)d_in[6];   // (2H,H)
    const float* b_g  = (const float*)d_in[7];   // (2H,)
    float* out = (float*)d_out;
    char*  ws  = (char*)d_ws;

    // Workspace layout (bytes)
    float*          Xp       = (float*)(ws);                      // 134217728
    float*          partials = (float*)(ws + 134217728ull);       //   8388608
    __hip_bfloat16* xbf      = (__hip_bfloat16*)(ws + 142606336ull); // 67108864
    __hip_bfloat16* wihb     = (__hip_bfloat16*)(ws + 209715200ull); //  2097152
    __hip_bfloat16* whhb     = (__hip_bfloat16*)(ws + 211812352ull); //  2097152
    __hip_bfloat16* wgb      = (__hip_bfloat16*)(ws + 213909504ull); //  4194304
    __hip_bfloat16* hbuf     = (__hip_bfloat16*)(ws + 218103808ull); //   131072
    __hip_bfloat16* hrnn     = (__hip_bfloat16*)(ws + 218234880ull); //   131072

    // Phase 0: bf16 conversions + h0 = 0
    k_f32_to_bf16<<<2048, 256, 0, stream>>>(x,    xbf,  Bb * Tt * Ii);
    k_f32_to_bf16<<<512,  256, 0, stream>>>(w_ih, wihb, Hh * Ii);
    k_f32_to_bf16<<<512,  256, 0, stream>>>(w_hh, whhb, Hh * Hh);
    k_f32_to_bf16<<<1024, 256, 0, stream>>>(w_g,  wgb,  2 * Hh * Hh);
    k_zero_bf16 <<<64,    256, 0, stream>>>(hbuf, Bb * Hh);

    // Phase 1: parallel input-projection GEMM (no sequential dependency)
    k_xp_gemm<<<dim3(Hh / 16, (Bb * Tt) / 64), 128, 0, stream>>>(
        xbf, wihb, b_ih, b_hh, Xp);

    // Phase 2: sequential recurrence, two WMMA kernels per timestep
    for (int t = 0; t < Tt; ++t) {
        k_rnn_step  <<<Hh / 16, 128, 0, stream>>>(hbuf, whhb, Xp, hrnn, t);
        k_gauss_step<<<Hh / 16, 128, 0, stream>>>(hrnn, wgb, b_g, eps,
                                                  out, hbuf, partials, t);
    }

    // Phase 3: deterministic log-prob reduction
    k_reduce_prob<<<(Tt * Bb + 255) / 256, 256, 0, stream>>>(partials, out);

    (void)in_sizes; (void)n_in; (void)out_size; (void)ws_size;
}
